// MCQTagger_60636348285046
// MI455X (gfx1250) — matrix operand order repaired and checked by reading, not compiled, not run on hardware
//
#include <hip/hip_runtime.h>
#include <hip/hip_bf16.h>
#include <math.h>

// ---------------------------------------------------------------------------
// BiLSTM-CRF tagger for MI455X (gfx1250, wave32, WMMA).
//   B=32, L=T=1024, IN0=290 (pad 320), HIDDEN=512, H=256, 4H=1024, NLAB=20 (pad 32)
// GEMMs use v_wmma_f32_16x16x32_f16 with branch-free K-loops (clamped-row
// loads instead of exec-masked guards). Recurrent scans are persistent
// 1-block kernels with h-state in LDS (128-bit ds loads) and per-wave
// gate-sliced WMMA so the LSTM nonlinearity needs no cross-lane traffic.
// ---------------------------------------------------------------------------

typedef __attribute__((ext_vector_type(16))) _Float16 v16h;
typedef __attribute__((ext_vector_type(8)))  _Float16 v8h;
typedef __attribute__((ext_vector_type(8)))  float    v8f;

union V16 { v16h v; v8h h[2]; _Float16 e[16]; };

#define TT 1024
#define BB 32
#define HH 256
#define G4 1024

static __device__ __forceinline__ float sigf(float x) { return 1.0f / (1.0f + expf(-x)); }

// ---------------------------------------------------------------------------
// fp32 -> fp16 convert with zero padding (rows_src x k_src) -> (rows_dst x k_dst)
// ---------------------------------------------------------------------------
__global__ void k_cvt_pad(const float* __restrict__ src, _Float16* __restrict__ dst,
                          int rows_src, int k_src, int rows_dst, int k_dst) {
    int idx = blockIdx.x * blockDim.x + threadIdx.x;
    int total = rows_dst * k_dst;
    if (idx >= total) return;
    int r = idx / k_dst, k = idx - r * k_dst;
    float v = (r < rows_src && k < k_src) ? src[r * k_src + k] : 0.0f;
    dst[idx] = (_Float16)v;
}

// ---------------------------------------------------------------------------
// Embedding assembly: word emb gather + char CNN(maxpool over relu) + bbox
// linear, written as fp16 rows [l*B + b] of width 320 (cols 290..319 zero).
// One wave per token; char embeds staged in LDS.
// ---------------------------------------------------------------------------
__global__ void k_embed(const int* __restrict__ words, const int* __restrict__ chars,
                        const float* __restrict__ bboxes,
                        const float* __restrict__ word_emb, const float* __restrict__ char_emb,
                        const float* __restrict__ w3, const float* __restrict__ b3,
                        const float* __restrict__ w4, const float* __restrict__ b4,
                        const float* __restrict__ w5, const float* __restrict__ b5,
                        const float* __restrict__ bbw, const float* __restrict__ bbb,
                        _Float16* __restrict__ x0h) {
    __shared__ float ce[4][30][16];       // per-wave char-emb tile [emb_dim][char_pos]
    const int lane = threadIdx.x & 31;
    const int wv   = threadIdx.x >> 5;
    const int tok  = blockIdx.x * 4 + wv; // token id = b*L + l
    const int b = tok >> 10, l = tok & 1023;
    const size_t row = (size_t)l * BB + b;
    _Float16* out = x0h + row * 320;

    // stage char embeddings: 16 chars x 30 dims
    for (int idx = lane; idx < 480; idx += 32) {
        int c = idx / 30, i = idx - c * 30;
        int ch = chars[(size_t)tok * 16 + c];
        ce[wv][i][c] = char_emb[ch * 30 + i];
    }
    __syncthreads();

    // word embedding (cols 0..99)
    {
        int w = words[tok];
        const float* we = word_emb + (size_t)w * 100;
        for (int i = lane; i < 100; i += 32) out[i] = (_Float16)we[i];
    }
    // char CNN (cols 100..189): 3 kernels x 30 out-channels, relu + maxpool
    for (int oidx = lane; oidx < 90; oidx += 32) {
        int k, o; const float *w, *bs;
        if (oidx < 30)      { k = 3; o = oidx;      w = w3; bs = b3; }
        else if (oidx < 60) { k = 4; o = oidx - 30; w = w4; bs = b4; }
        else                { k = 5; o = oidx - 60; w = w5; bs = b5; }
        float best = 0.0f;                 // relu(.) >= 0 so max >= 0
        for (int p = 0; p <= 16 - k; ++p) {
            float s = bs[o];
            for (int i = 0; i < 30; ++i)
                for (int t = 0; t < k; ++t)
                    s += ce[wv][i][p + t] * w[(o * 30 + i) * k + t];
            best = fmaxf(best, s);
        }
        out[100 + oidx] = (_Float16)best;
    }
    // bbox linear (cols 190..289)
    {
        float q0 = bboxes[(size_t)tok * 4 + 0], q1 = bboxes[(size_t)tok * 4 + 1];
        float q2 = bboxes[(size_t)tok * 4 + 2], q3 = bboxes[(size_t)tok * 4 + 3];
        for (int o = lane; o < 100; o += 32) {
            float s = bbb[o] + q0 * bbw[o * 4] + q1 * bbw[o * 4 + 1]
                             + q2 * bbw[o * 4 + 2] + q3 * bbw[o * 4 + 3];
            out[190 + o] = (_Float16)s;
        }
    }
    // K padding (cols 290..319)
    for (int i = lane; i < 30; i += 32) out[290 + i] = (_Float16)0.0f;
}

// ---------------------------------------------------------------------------
// Generic WMMA GEMM:  C[M,N] = A[M,K](f16) * W[N,K](f16)^T + bias
// Block tile 128x64, 8 waves (4M x 2N), each wave 32x32 via 2x2 WMMA accs.
// Branch-free K loop: out-of-range N columns load a clamped (valid) row and
// are discarded by the store-side guard -> no exec-mask churn in the hot loop.
// ---------------------------------------------------------------------------
__global__ void k_gemm_wmma(const _Float16* __restrict__ A, int lda,
                            const _Float16* __restrict__ W, int ldw,
                            const float* __restrict__ bias, int biasN,
                            float* __restrict__ C, int ldc,
                            int N, int K) {
    const int lane = threadIdx.x & 31;
    const int wave = threadIdx.x >> 5;
    const int mb = blockIdx.y * 128 + (wave & 3) * 32;   // wave row base
    const int nb = blockIdx.x * 64  + (wave >> 2) * 32;  // wave col base
    const int l15 = lane & 15, lhi = lane >> 4;

    // clamped W row indices (loop-invariant, branch-free)
    int nrow[2];
    #pragma unroll
    for (int nt = 0; nt < 2; ++nt) {
        int n = nb + 16 * nt + l15;
        nrow[nt] = (n < N) ? n : 0;
    }
    const _Float16* aprow[2];
    #pragma unroll
    for (int mt = 0; mt < 2; ++mt)
        aprow[mt] = A + (size_t)(mb + 16 * mt + l15) * lda + lhi * 8;
    const _Float16* wprow[2];
    #pragma unroll
    for (int nt = 0; nt < 2; ++nt)
        wprow[nt] = W + (size_t)nrow[nt] * ldw + lhi * 8;

    v8f acc[2][2];
    { v8f z = {}; acc[0][0] = z; acc[0][1] = z; acc[1][0] = z; acc[1][1] = z; }

    const int ksteps = K >> 5;
    for (int ks = 0; ks < ksteps; ++ks) {
        const int kb = ks << 5;
        // A fragments (16x32 f16): lane<16 -> K {0..7,16..23}, lane>=16 -> +8
        V16 a[2];
        #pragma unroll
        for (int mt = 0; mt < 2; ++mt) {
            a[mt].h[0] = *(const v8h*)(aprow[mt] + kb);
            a[mt].h[1] = *(const v8h*)(aprow[mt] + kb + 16);
        }
        if (ks + 1 < ksteps)
            __builtin_prefetch(aprow[0] + kb + 32, 0, 1);
        // W fragments (B matrix 32x16): lane<16 -> K {0..7,16..23}, lane>=16 -> +8
        #pragma unroll
        for (int nt = 0; nt < 2; ++nt) {
            V16 bw;
            bw.h[0] = *(const v8h*)(wprow[nt] + kb);
            bw.h[1] = *(const v8h*)(wprow[nt] + kb + 16);
            #pragma unroll
            for (int mt = 0; mt < 2; ++mt)
                acc[mt][nt] = __builtin_amdgcn_wmma_f32_16x16x32_f16(
                    false, a[mt].v, false, bw.v, (short)0, acc[mt][nt], false, false);
        }
    }
    // epilogue: bias add + guarded store
    #pragma unroll
    for (int mt = 0; mt < 2; ++mt)
        #pragma unroll
        for (int nt = 0; nt < 2; ++nt)
            #pragma unroll
            for (int r = 0; r < 8; ++r) {
                int row = mb + 16 * mt + r + 8 * lhi;
                int col = nb + 16 * nt + l15;
                if (col < N)
                    C[(size_t)row * ldc + col] =
                        acc[mt][nt][r] + (col < biasN ? bias[col] : 0.0f);
            }
}

// ---------------------------------------------------------------------------
// Persistent LSTM scan (one workgroup, 512 threads = 16 waves).
// pre: [T][B][1024] f32 pre-activations (x@Wih^T + b), whh: [1024][256] f16.
// Wave w owns hidden slice h = w*16 + lane%16 across all 4 gates -> i/f/g/o of
// the same (b,h) land in the same lane/slot; c kept in registers, h in LDS.
// LDS row stride 264 halves keeps every fragment 16B-aligned (ds_load_b128).
// ---------------------------------------------------------------------------
__global__ void __launch_bounds__(512)
k_lstm_scan(const float* __restrict__ pre, const _Float16* __restrict__ whh,
            _Float16* __restrict__ xout, int out_ld, int out_off, int rev) {
    __shared__ _Float16 h_sh[BB][HH + 8];
    const int lane = threadIdx.x & 31;
    const int w    = threadIdx.x >> 5;          // 0..15
    const int l15  = lane & 15, lhi = lane >> 4;
    const int hidx = w * 16 + l15;

    for (int i = threadIdx.x; i < BB * (HH + 8); i += 512)
        (&h_sh[0][0])[i] = (_Float16)0.0f;
    __syncthreads();

    // loop-invariant W pointers: 4 gates for this wave's hidden slice
    const _Float16* wgp[4];
    #pragma unroll
    for (int g = 0; g < 4; ++g)
        wgp[g] = whh + (size_t)(g * HH + hidx) * HH + lhi * 8;

    float c[16];
    #pragma unroll
    for (int i = 0; i < 16; ++i) c[i] = 0.0f;

    for (int s = 0; s < TT; ++s) {
        const int t = rev ? (TT - 1 - s) : s;
        v8f acc[4][2];
        { v8f z = {};
          for (int g = 0; g < 4; ++g) { acc[g][0] = z; acc[g][1] = z; } }

        // g += h @ Whh^T   (K=256, 8 WMMA K-steps)
        #pragma unroll 2
        for (int ks = 0; ks < 8; ++ks) {
            const int kb = ks << 5;
            V16 a[2];
            #pragma unroll
            for (int mt = 0; mt < 2; ++mt) {
                const _Float16* hp = &h_sh[16 * mt + l15][kb + lhi * 8];
                a[mt].h[0] = *(const v8h*)(hp);
                a[mt].h[1] = *(const v8h*)(hp + 16);
            }
            #pragma unroll
            for (int g = 0; g < 4; ++g) {
                V16 bw;
                bw.h[0] = *(const v8h*)(wgp[g] + kb);
                bw.h[1] = *(const v8h*)(wgp[g] + kb + 16);
                acc[g][0] = __builtin_amdgcn_wmma_f32_16x16x32_f16(
                    false, a[0].v, false, bw.v, (short)0, acc[g][0], false, false);
                acc[g][1] = __builtin_amdgcn_wmma_f32_16x16x32_f16(
                    false, a[1].v, false, bw.v, (short)0, acc[g][1], false, false);
            }
        }

        // gate nonlinearities (per-lane: all 4 gates share (mt,r) slots)
        float hval[16];
        const float* pt = pre + (size_t)t * BB * G4;
        #pragma unroll
        for (int mt = 0; mt < 2; ++mt)
            #pragma unroll
            for (int r = 0; r < 8; ++r) {
                const int b = 16 * mt + r + 8 * lhi;
                const float* pb = pt + (size_t)b * G4;
                float gi = acc[0][mt][r] + pb[0 * HH + hidx];
                float gf = acc[1][mt][r] + pb[1 * HH + hidx];
                float gg = acc[2][mt][r] + pb[2 * HH + hidx];
                float go = acc[3][mt][r] + pb[3 * HH + hidx];
                float cn = sigf(gf) * c[mt * 8 + r] + sigf(gi) * tanhf(gg);
                c[mt * 8 + r] = cn;
                hval[mt * 8 + r] = sigf(go) * tanhf(cn);
            }

        __syncthreads();   // all waves done reading old h
        #pragma unroll
        for (int mt = 0; mt < 2; ++mt)
            #pragma unroll
            for (int r = 0; r < 8; ++r) {
                const int b = 16 * mt + r + 8 * lhi;
                _Float16 hv = (_Float16)hval[mt * 8 + r];
                h_sh[b][hidx] = hv;
                xout[((size_t)t * BB + b) * out_ld + out_off + hidx] = hv;
            }
        __syncthreads();   // new h visible
    }
}

// ---------------------------------------------------------------------------
// CRF: forward algorithm (logZ) + gold-path score, loss = sum_b(logZ - score).
// em: [T*B][32] f32 (cols >=20 unused). Single block, alpha/trans in LDS.
// ---------------------------------------------------------------------------
__global__ void k_crf(const float* __restrict__ em, const int* __restrict__ labels,
                      const unsigned char* __restrict__ mask,
                      const float* __restrict__ start, const float* __restrict__ endv,
                      const float* __restrict__ trans, float* __restrict__ out) {
    __shared__ float alpha[BB][20];
    __shared__ float tr[20][20];
    __shared__ float diff[BB];
    const int tid = threadIdx.x;
    if (tid < 400) tr[tid / 20][tid % 20] = trans[tid];
    const bool active = tid < BB * 20;
    const int b = tid / 20, j = tid - b * 20;
    if (active) alpha[b][j] = start[j] + em[(size_t)b * 32 + j];
    __syncthreads();

    for (int t = 1; t < TT; ++t) {
        float nv = 0.0f;
        if (active) {
            float mx = -1e30f;
            #pragma unroll
            for (int i = 0; i < 20; ++i) mx = fmaxf(mx, alpha[b][i] + tr[i][j]);
            float sm = 0.0f;
            #pragma unroll
            for (int i = 0; i < 20; ++i) sm += expf(alpha[b][i] + tr[i][j] - mx);
            float nxt = mx + logf(sm) + em[((size_t)t * BB + b) * 32 + j];
            nv = mask[b * TT + t] ? nxt : alpha[b][j];
        }
        __syncthreads();
        if (active) alpha[b][j] = nv;
        __syncthreads();
    }

    if (tid < BB) {
        // logZ
        float mx = -1e30f;
        for (int i = 0; i < 20; ++i) mx = fmaxf(mx, alpha[tid][i] + endv[i]);
        float sm = 0.0f;
        for (int i = 0; i < 20; ++i) sm += expf(alpha[tid][i] + endv[i] - mx);
        float lz = mx + logf(sm);
        // gold score
        int prev = labels[tid * TT];
        float sc = start[prev] + em[(size_t)tid * 32 + prev];
        for (int t = 1; t < TT; ++t) {
            int tg = labels[tid * TT + t];
            float mf = mask[tid * TT + t] ? 1.0f : 0.0f;
            sc += (tr[prev][tg] + em[((size_t)t * BB + tid) * 32 + tg]) * mf;
            prev = (mf > 0.0f) ? tg : prev;
        }
        sc += endv[prev];
        diff[tid] = lz - sc;
    }
    __syncthreads();
    if (tid == 0) {
        float s = 0.0f;
        for (int i = 0; i < BB; ++i) s += diff[i];
        out[0] = s;
    }
}

// ---------------------------------------------------------------------------
extern "C" void kernel_launch(void* const* d_in, const int* in_sizes, int n_in,
                              void* d_out, int out_size, void* d_ws, size_t ws_size,
                              hipStream_t stream) {
    (void)n_in; (void)out_size; (void)ws_size;
    // ---- inputs -----------------------------------------------------------
    const int*   words   = (const int*)d_in[0];
    const int*   chars   = (const int*)d_in[1];
    const float* bboxes  = (const float*)d_in[2];
    const unsigned char* mask = (const unsigned char*)d_in[3];
    const int*   labels  = (const int*)d_in[4];
    const float* wemb    = (const float*)d_in[5];
    const float* cemb    = (const float*)d_in[6];
    const float* w3 = (const float*)d_in[7],  *b3 = (const float*)d_in[8];
    const float* w4 = (const float*)d_in[9],  *b4 = (const float*)d_in[10];
    const float* w5 = (const float*)d_in[11], *b5 = (const float*)d_in[12];
    const float* bbw = (const float*)d_in[13], *bbb = (const float*)d_in[14];
    // lstm_params dict flattening order: detect insertion vs sorted-key order.
    // insertion: wih_0f whh_0f b_0f wih_0b whh_0b b_0b wih_1f whh_1f b_1f wih_1b whh_1b b_1b
    int I[12];
    if (in_sizes[15] == 1024 * 290) {
        for (int i = 0; i < 12; ++i) I[i] = 15 + i;
    } else { // sorted: b_0b b_0f b_1b b_1f whh_0b whh_0f whh_1b whh_1f wih_0b wih_0f wih_1b wih_1f
        I[0] = 24; I[1] = 20; I[2] = 16; I[3] = 23; I[4] = 19; I[5] = 15;
        I[6] = 26; I[7] = 22; I[8] = 18; I[9] = 25; I[10] = 21; I[11] = 17;
    }
    const float* wih0f = (const float*)d_in[I[0]];
    const float* whh0f = (const float*)d_in[I[1]];
    const float* bi0f  = (const float*)d_in[I[2]];
    const float* wih0b = (const float*)d_in[I[3]];
    const float* whh0b = (const float*)d_in[I[4]];
    const float* bi0b  = (const float*)d_in[I[5]];
    const float* wih1f = (const float*)d_in[I[6]];
    const float* whh1f = (const float*)d_in[I[7]];
    const float* bi1f  = (const float*)d_in[I[8]];
    const float* wih1b = (const float*)d_in[I[9]];
    const float* whh1b = (const float*)d_in[I[10]];
    const float* bi1b  = (const float*)d_in[I[11]];
    const float* ffw = (const float*)d_in[27], *ffb = (const float*)d_in[28];
    const float* crf_s = (const float*)d_in[29], *crf_e = (const float*)d_in[30];
    const float* crf_t = (const float*)d_in[31];

    // ---- workspace layout -------------------------------------------------
    char* ws = (char*)d_ws;
    size_t cur = 0;
    auto wsalloc = [&](size_t bytes) -> char* {
        char* p = ws + cur;
        cur += (bytes + 255) & ~(size_t)255;
        return p;
    };
    const size_t MB_rows = (size_t)TT * BB;                      // 32768
    _Float16* x0h    = (_Float16*)wsalloc(MB_rows * 320 * 2);
    _Float16* x1h    = (_Float16*)wsalloc(MB_rows * 512 * 2);
    _Float16* x2h    = (_Float16*)wsalloc(MB_rows * 512 * 2);
    _Float16* wih0fh = (_Float16*)wsalloc(1024 * 320 * 2);
    _Float16* wih0bh = (_Float16*)wsalloc(1024 * 320 * 2);
    _Float16* wih1fh = (_Float16*)wsalloc(1024 * 512 * 2);
    _Float16* wih1bh = (_Float16*)wsalloc(1024 * 512 * 2);
    _Float16* whh0fh = (_Float16*)wsalloc(1024 * 256 * 2);
    _Float16* whh0bh = (_Float16*)wsalloc(1024 * 256 * 2);
    _Float16* whh1fh = (_Float16*)wsalloc(1024 * 256 * 2);
    _Float16* whh1bh = (_Float16*)wsalloc(1024 * 256 * 2);
    _Float16* ffwh   = (_Float16*)wsalloc(32 * 512 * 2);
    float*    pre    = (float*)wsalloc(MB_rows * (size_t)G4 * 4); // reused per scan
    float*    em     = (float*)wsalloc(MB_rows * 32 * 4);

    // ---- weight conversion ------------------------------------------------
    auto cvt = [&](const float* src, _Float16* dst, int rs, int ks, int rd, int kd) {
        int total = rd * kd;
        k_cvt_pad<<<(total + 255) / 256, 256, 0, stream>>>(src, dst, rs, ks, rd, kd);
    };
    cvt(wih0f, wih0fh, 1024, 290, 1024, 320);
    cvt(wih0b, wih0bh, 1024, 290, 1024, 320);
    cvt(wih1f, wih1fh, 1024, 512, 1024, 512);
    cvt(wih1b, wih1bh, 1024, 512, 1024, 512);
    cvt(whh0f, whh0fh, 1024, 256, 1024, 256);
    cvt(whh0b, whh0bh, 1024, 256, 1024, 256);
    cvt(whh1f, whh1fh, 1024, 256, 1024, 256);
    cvt(whh1b, whh1bh, 1024, 256, 1024, 256);
    cvt(ffw,   ffwh,   20,   512, 32,   512);

    // ---- embedding assembly ----------------------------------------------
    k_embed<<<(BB * TT) / 4, 128, 0, stream>>>(words, chars, bboxes, wemb, cemb,
                                               w3, b3, w4, b4, w5, b5, bbw, bbb, x0h);

    // ---- GEMM + scan pipeline --------------------------------------------
    auto gemm = [&](const _Float16* A, int lda, const _Float16* W, int ldw,
                    const float* bias, int biasN, float* C, int ldc, int N, int K) {
        dim3 grid((N + 63) / 64, (unsigned)(MB_rows / 128));
        k_gemm_wmma<<<grid, 256, 0, stream>>>(A, lda, W, ldw, bias, biasN, C, ldc, N, K);
    };
    auto scan = [&](const _Float16* whh_h, _Float16* xo, int off, int rev) {
        k_lstm_scan<<<1, 512, 0, stream>>>(pre, whh_h, xo, 512, off, rev);
    };
    // layer 0
    gemm(x0h, 320, wih0fh, 320, bi0f, 1024, pre, G4, 1024, 320);
    scan(whh0fh, x1h, 0, 0);
    gemm(x0h, 320, wih0bh, 320, bi0b, 1024, pre, G4, 1024, 320);
    scan(whh0bh, x1h, 256, 1);
    // layer 1
    gemm(x1h, 512, wih1fh, 512, bi1f, 1024, pre, G4, 1024, 512);
    scan(whh1fh, x2h, 0, 0);
    gemm(x1h, 512, wih1bh, 512, bi1b, 1024, pre, G4, 1024, 512);
    scan(whh1bh, x2h, 256, 1);
    // emissions: [32768,512] x [32,512]^T  (cols 20..31 are zero-weight pads)
    gemm(x2h, 512, ffwh, 512, ffb, 20, em, 32, 32, 512);

    // ---- CRF loss ---------------------------------------------------------
    k_crf<<<1, 1024, 0, stream>>>(em, labels, mask, crf_s, crf_e, crf_t, (float*)d_out);
}
